// GroupedQueryAttention_53446573032007
// MI455X (gfx1250) — matrix operand order repaired and checked by reading, not compiled, not run on hardware
//
#include <hip/hip_runtime.h>
#include <hip/hip_bf16.h>
#include <stdint.h>

// ---------------------------------------------------------------------------
// GQA for MI455X (gfx1250): all matmuls via v_wmma_f32_16x16x32_bf16.
// B=2, S=2048, D=2048, H=16, HKV=4, DH=128, NREP=4.
// ---------------------------------------------------------------------------

typedef __attribute__((ext_vector_type(16))) __bf16 v16bf;
typedef __attribute__((ext_vector_type(8)))  float  v8f;

#define NHEAD   16
#define NKV     4
#define DHEAD   128
#define SEQ     2048
#define BATCH   2
#define DMODEL  2048
#define KVW     (NKV * DHEAD)   // 512

__device__ __forceinline__ unsigned short f32_to_bf16(float f) {
  union { float f; uint32_t u; } c; c.f = f;
  uint32_t u = c.u;
  u += 0x7FFFu + ((u >> 16) & 1u);          // round-to-nearest-even
  return (unsigned short)(u >> 16);
}

// A-operand fragment (16x32 bf16, M x K) from a row-major bf16 buffer.
// Layout: lanes 0-15 hold row M=lane, K=0..15;
// lanes 16-31 hold row M=lane-16, K=16..31.
__device__ __forceinline__ v16bf load_a16x32(const unsigned short* base,
                                             int rowStride, int lane) {
  const int r  = lane & 15;
  const int k0 = (lane >> 4) << 4;          // 0 or 16
  const unsigned short* p = base + (size_t)r * rowStride + k0;
  union { v16bf v; uint4 q[2]; } u;
  u.q[0] = *(const uint4*)(p);              // K = k0 .. k0+7
  u.q[1] = *(const uint4*)(p + 8);          // K = k0+8 .. k0+15
  return u.v;
}

// B-operand fragment (32x16 bf16, K x N) read from a transposed (N-major)
// buffer BT[n][k]: lane<16 reads BT[n=lane][k=0..15], lane>=16 reads
// BT[n=lane-16][k=16..31]; VGPR v holds rows k=2v,2v+1 packed lo/hi.
__device__ __forceinline__ v16bf load_bt32x16(const unsigned short* base,
                                              int rowStride, int lane) {
  const int n    = lane & 15;
  const int koff = (lane >> 4) << 4;        // 0 or 16
  const unsigned short* p = base + (size_t)n * rowStride + koff;
  union { v16bf v; uint4 q[2]; } u;
  u.q[0] = *(const uint4*)(p);              // k = koff .. koff+7
  u.q[1] = *(const uint4*)(p + 8);          // k = koff+8 .. koff+15
  return u.v;
}

// ---------------------------------------------------------------------------
// GEMM: C[M,N] = A[M,K] @ B[K,N].  A is f32 or bf16 (converted to bf16 while
// staging to LDS); B is f32 (weights) staged transposed as bf16; C is bf16 or
// f32.  Block = 256 threads (8 waves), tile 128x128, K-step 32, wave tile
// 32x64 -> 8 WMMAs per wave per K-step.
// ---------------------------------------------------------------------------
#define BM 128
#define BN 128
#define BK 32
#define ASTR 48   // LDS row stride (shorts), 96B: 16B-aligned, bank-spread
#define BSTR 48

template <bool A_F32, bool OUT_BF16>
__global__ __launch_bounds__(256) void gemm_wmma_bf16(
    const void* __restrict__ Ap, const float* __restrict__ Bp,
    void* __restrict__ Cp, int M, int N, int K) {
  __shared__ unsigned short As[BM * ASTR];
  __shared__ unsigned short Bs[BN * BSTR];

  const int t = threadIdx.x;
  const int wave = t >> 5, lane = t & 31;
  const int m0 = blockIdx.y * BM;
  const int n0 = blockIdx.x * BN;
  const int wm = (wave >> 1) * 32;          // 0,32,64,96
  const int wn = (wave & 1) * 64;           // 0,64

  const v8f vzero = {0.f, 0.f, 0.f, 0.f, 0.f, 0.f, 0.f, 0.f};
  v8f acc[2][4];
  #pragma unroll
  for (int i = 0; i < 2; ++i)
    #pragma unroll
    for (int j = 0; j < 4; ++j) acc[i][j] = vzero;

  // staging indices: A -> each thread moves one 16-element half-row
  const int ar = t >> 1;                    // 0..127
  const int ac = (t & 1) * 16;              // 0 or 16
  // B -> each thread moves a 16-deep column slice into BT
  const int bn = t >> 1;                    // 0..127
  const int bk = (t & 1) * 16;              // 0 or 16

  for (int k0 = 0; k0 < K; k0 += BK) {
    __syncthreads();
    // ---- stage A tile (convert f32 -> bf16 if needed) ----
    if constexpr (A_F32) {
      const float* Ag = (const float*)Ap + (size_t)(m0 + ar) * K + k0 + ac;
      float fv[16];
      #pragma unroll
      for (int j = 0; j < 4; ++j) {
        float4 f = ((const float4*)Ag)[j];
        fv[4 * j + 0] = f.x; fv[4 * j + 1] = f.y;
        fv[4 * j + 2] = f.z; fv[4 * j + 3] = f.w;
      }
      uint32_t w[8];
      #pragma unroll
      for (int j = 0; j < 8; ++j)
        w[j] = (uint32_t)f32_to_bf16(fv[2 * j]) |
               ((uint32_t)f32_to_bf16(fv[2 * j + 1]) << 16);
      uint4* dst = (uint4*)&As[ar * ASTR + ac];
      dst[0] = make_uint4(w[0], w[1], w[2], w[3]);
      dst[1] = make_uint4(w[4], w[5], w[6], w[7]);
    } else {
      const uint4* Ag = (const uint4*)((const unsigned short*)Ap +
                                       (size_t)(m0 + ar) * K + k0 + ac);
      uint4* dst = (uint4*)&As[ar * ASTR + ac];
      dst[0] = Ag[0];
      dst[1] = Ag[1];
    }
    // prefetch next A K-tile (global_prefetch_b8)
    if (k0 + BK < K) {
      const char* np =
          A_F32 ? (const char*)((const float*)Ap +
                                (size_t)(m0 + ar) * K + (k0 + BK) + ac)
                : (const char*)((const unsigned short*)Ap +
                                (size_t)(m0 + ar) * K + (k0 + BK) + ac);
      __builtin_prefetch(np, 0, 1);
    }
    // ---- stage B tile transposed: Bs[n][k] = bf16(B[k0+k][n0+n]) ----
    {
      const float* Bg = Bp + (size_t)(k0 + bk) * N + (n0 + bn);
      uint32_t w[8];
      #pragma unroll
      for (int j = 0; j < 8; ++j) {
        float lo = Bg[(size_t)(2 * j) * N];
        float hi = Bg[(size_t)(2 * j + 1) * N];
        w[j] = (uint32_t)f32_to_bf16(lo) | ((uint32_t)f32_to_bf16(hi) << 16);
      }
      uint4* dst = (uint4*)&Bs[bn * BSTR + bk];
      dst[0] = make_uint4(w[0], w[1], w[2], w[3]);
      dst[1] = make_uint4(w[4], w[5], w[6], w[7]);
    }
    __syncthreads();

    // ---- compute: 8 WMMAs per wave per K-step ----
    v16bf a0 = load_a16x32(&As[(wm + 0) * ASTR], ASTR, lane);
    v16bf a1 = load_a16x32(&As[(wm + 16) * ASTR], ASTR, lane);
    #pragma unroll
    for (int j = 0; j < 4; ++j) {
      v16bf bj = load_bt32x16(&Bs[(wn + j * 16) * BSTR], BSTR, lane);
      acc[0][j] = __builtin_amdgcn_wmma_f32_16x16x32_bf16(
          false, a0, false, bj, (short)0, acc[0][j], false, false);
      acc[1][j] = __builtin_amdgcn_wmma_f32_16x16x32_bf16(
          false, a1, false, bj, (short)0, acc[1][j], false, false);
    }
  }

  // ---- store C (f32 C/D layout: VGPR r -> row r (+8 for upper lanes)) ----
  const int hf = lane >> 4;
  const int ln = lane & 15;
  #pragma unroll
  for (int i = 0; i < 2; ++i) {
    #pragma unroll
    for (int j = 0; j < 4; ++j) {
      const int row0 = m0 + wm + i * 16 + hf * 8;
      const int col = n0 + wn + j * 16 + ln;
      #pragma unroll
      for (int r = 0; r < 8; ++r) {
        float v = acc[i][j][r];
        if constexpr (OUT_BF16)
          ((unsigned short*)Cp)[(size_t)(row0 + r) * N + col] = f32_to_bf16(v);
        else
          ((float*)Cp)[(size_t)(row0 + r) * N + col] = v;
      }
    }
  }
}

// ---------------------------------------------------------------------------
// Flash attention (GQA): grid (B*H, S/64), block 128 (4 waves).
// Each wave owns a 16-row query tile, full 16x128 f32 O accumulator, and
// per-row online-softmax stats. 16 WMMAs per wave per 32-key step.
// K tile is staged with gfx1250 async global->LDS copies (ASYNCcnt).
// ---------------------------------------------------------------------------
#define KSTR 136   // 32x128 K tile rows, 272B stride (16B aligned)
#define VSTR 48    // 128x32 V^T tile
#define PSTR 48    // per-wave 16x32 P tile

__global__ __launch_bounds__(128) void gqa_flash_wmma(
    const unsigned short* __restrict__ Q, const unsigned short* __restrict__ Kc,
    const unsigned short* __restrict__ Vc, const int* __restrict__ mask,
    unsigned short* __restrict__ O) {
  __shared__ unsigned short Ks[32 * KSTR];
  __shared__ unsigned short Vt[128 * VSTR];
  __shared__ unsigned short Ps[4 * 16 * PSTR];

  const int t = threadIdx.x;
  const int wave = t >> 5, lane = t & 31;
  const int b = blockIdx.x / NHEAD;
  const int h = blockIdx.x % NHEAD;
  const int g = h >> 2;                      // kv head (NREP=4)
  const int qbase = blockIdx.y * 64 + wave * 16;
  const int hf = lane >> 4;
  const int ln = lane & 15;

  // Q fragments: 16 rows x 128 dh as 4 A-frags, straight from global (L2-hot)
  const unsigned short* Qg =
      Q + (size_t)(b * SEQ + qbase) * DMODEL + h * DHEAD;
  v16bf aq[4];
  #pragma unroll
  for (int c = 0; c < 4; ++c) aq[c] = load_a16x32(Qg + c * 32, DMODEL, lane);

  const v8f vzero = {0.f, 0.f, 0.f, 0.f, 0.f, 0.f, 0.f, 0.f};
  v8f oacc[8];
  float mstate[8], lstate[8];
  #pragma unroll
  for (int r = 0; r < 8; ++r) {
    oacc[r] = vzero;
    mstate[r] = -3.0e38f;
    lstate[r] = 0.f;
  }
  const float scale = 0.0883883476483184f;   // 1/sqrt(128)

  const int skey = t >> 2;                   // 0..31
  const int scc = (t & 3) * 32;              // 0..96
  const int vdh = t;                         // 0..127

  for (int k0 = 0; k0 < SEQ; k0 += 32) {
    __syncthreads();
    // stage K rows via async global->LDS (64B per thread, 4x b128).
    // INST_OFFSET applies to both the LDS and global addresses.
    {
      const uint64_t gaddr = (uint64_t)(uintptr_t)(
          Kc + (size_t)(b * SEQ + k0 + skey) * KVW + g * DHEAD + scc);
      const uint32_t laddr =
          (uint32_t)(uintptr_t)&Ks[skey * KSTR + scc];  // LDS byte offset
      asm volatile(
          "global_load_async_to_lds_b128 %0, %1, off\n\t"
          "global_load_async_to_lds_b128 %0, %1, off offset:16\n\t"
          "global_load_async_to_lds_b128 %0, %1, off offset:32\n\t"
          "global_load_async_to_lds_b128 %0, %1, off offset:48"
          :
          : "v"(laddr), "v"(gaddr)
          : "memory");
    }
    // stage V transposed: Vt[dh][key]
    {
      const unsigned short* src =
          Vc + (size_t)(b * SEQ + k0) * KVW + g * DHEAD + vdh;
      unsigned short* dst = &Vt[vdh * VSTR];
      #pragma unroll
      for (int kk = 0; kk < 32; ++kk) dst[kk] = src[(size_t)kk * KVW];
    }
    asm volatile("s_wait_asynccnt 0x0" ::: "memory");
    __syncthreads();

    // scores: S = Q . K^T  (K-dim = DH, 4 chunks of 32; 2 key tiles)
    v8f s0 = vzero, s1 = vzero;
    #pragma unroll
    for (int c = 0; c < 4; ++c) {
      v16bf bk0 = load_bt32x16(&Ks[0 * KSTR + c * 32], KSTR, lane);
      v16bf bk1 = load_bt32x16(&Ks[16 * KSTR + c * 32], KSTR, lane);
      s0 = __builtin_amdgcn_wmma_f32_16x16x32_bf16(false, aq[c], false, bk0,
                                                   (short)0, s0, false, false);
      s1 = __builtin_amdgcn_wmma_f32_16x16x32_bf16(false, aq[c], false, bk1,
                                                   (short)0, s1, false, false);
    }

    // scale + mask
    {
      const size_t mbase = (size_t)b * SEQ * SEQ +
                           (size_t)(qbase + hf * 8) * SEQ + k0 + ln;
      #pragma unroll
      for (int r = 0; r < 8; ++r) {
        const size_t mi = mbase + (size_t)r * SEQ;
        float v0 = s0[r] * scale;
        float v1 = s1[r] * scale;
        s0[r] = (mask[mi] == 0) ? -1.0e9f : v0;
        s1[r] = (mask[mi + 16] == 0) ? -1.0e9f : v1;
      }
    }

    // online softmax: row = r + 8*hf lives across the lane's 16-lane half
    #pragma unroll
    for (int r = 0; r < 8; ++r) {
      float mx = fmaxf(s0[r], s1[r]);
      mx = fmaxf(mx, __shfl_xor(mx, 1, 32));
      mx = fmaxf(mx, __shfl_xor(mx, 2, 32));
      mx = fmaxf(mx, __shfl_xor(mx, 4, 32));
      mx = fmaxf(mx, __shfl_xor(mx, 8, 32));
      float mnew = fmaxf(mstate[r], mx);
      float corr = __expf(mstate[r] - mnew);
      float p0 = __expf(s0[r] - mnew);
      float p1 = __expf(s1[r] - mnew);
      s0[r] = p0;
      s1[r] = p1;
      float sm = p0 + p1;
      sm += __shfl_xor(sm, 1, 32);
      sm += __shfl_xor(sm, 2, 32);
      sm += __shfl_xor(sm, 4, 32);
      sm += __shfl_xor(sm, 8, 32);
      lstate[r] = lstate[r] * corr + sm;
      mstate[r] = mnew;
      #pragma unroll
      for (int d = 0; d < 8; ++d) oacc[d][r] *= corr;
    }

    // P (C-layout regs) -> LDS -> A-layout fragment
    unsigned short* Pw = &Ps[wave * 16 * PSTR];
    #pragma unroll
    for (int r = 0; r < 8; ++r) {
      Pw[(r + hf * 8) * PSTR + ln] = f32_to_bf16(s0[r]);
      Pw[(r + hf * 8) * PSTR + 16 + ln] = f32_to_bf16(s1[r]);
    }
    __syncthreads();  // make cross-lane LDS writes visible before A-frag read
    v16bf ap = load_a16x32(Pw, PSTR, lane);

    // O += P . V  (8 dh tiles, K-dim = 32 keys)
    #pragma unroll
    for (int d = 0; d < 8; ++d) {
      v16bf bv = load_bt32x16(&Vt[(d * 16) * VSTR], VSTR, lane);
      oacc[d] = __builtin_amdgcn_wmma_f32_16x16x32_bf16(
          false, ap, false, bv, (short)0, oacc[d], false, false);
    }
  }

  // epilogue: O /= l, write bf16 to attn-out [B,S,D]
  unsigned short* Og =
      O + (size_t)(b * SEQ + qbase + hf * 8) * DMODEL + h * DHEAD + ln;
  #pragma unroll
  for (int r = 0; r < 8; ++r) {
    float inv = 1.0f / fmaxf(lstate[r], 1e-30f);
    #pragma unroll
    for (int d = 0; d < 8; ++d)
      Og[(size_t)r * DMODEL + d * 16] = f32_to_bf16(oacc[d][r] * inv);
  }
}

// ---------------------------------------------------------------------------
// Launch
// ---------------------------------------------------------------------------
extern "C" void kernel_launch(void* const* d_in, const int* in_sizes, int n_in,
                              void* d_out, int out_size, void* d_ws,
                              size_t ws_size, hipStream_t stream) {
  const float* query = (const float*)d_in[0];
  const float* key   = (const float*)d_in[1];
  const float* value = (const float*)d_in[2];
  const int*   mask  = (const int*)d_in[3];
  const float* Wq    = (const float*)d_in[4];
  const float* Wk    = (const float*)d_in[5];
  const float* Wv    = (const float*)d_in[6];
  const float* Wo    = (const float*)d_in[7];
  float* out = (float*)d_out;

  const size_t M = (size_t)BATCH * SEQ;      // 4096
  char* w = (char*)d_ws;
  unsigned short* Qb = (unsigned short*)(w);                       // M x 2048
  unsigned short* Kb = (unsigned short*)(w + M * DMODEL * 2);      // M x 512
  unsigned short* Vb = (unsigned short*)(w + M * DMODEL * 2 + M * KVW * 2);
  unsigned short* Ab = (unsigned short*)(w + M * DMODEL * 2 + 2 * M * KVW * 2);

  dim3 blk(256);
  // Q/K/V projections (f32 in, bf16 out)
  gemm_wmma_bf16<true, true><<<dim3(DMODEL / BN, M / BM), blk, 0, stream>>>(
      query, Wq, Qb, (int)M, DMODEL, DMODEL);
  gemm_wmma_bf16<true, true><<<dim3(KVW / BN, M / BM), blk, 0, stream>>>(
      key, Wk, Kb, (int)M, KVW, DMODEL);
  gemm_wmma_bf16<true, true><<<dim3(KVW / BN, M / BM), blk, 0, stream>>>(
      value, Wv, Vb, (int)M, KVW, DMODEL);

  // attention
  gqa_flash_wmma<<<dim3(BATCH * NHEAD, SEQ / 64), dim3(128), 0, stream>>>(
      Qb, Kb, Vb, mask, Ab);

  // output projection (bf16 in, f32 out)
  gemm_wmma_bf16<false, false><<<dim3(DMODEL / BN, M / BM), blk, 0, stream>>>(
      Ab, Wo, out, (int)M, DMODEL, DMODEL);
}
